// CrossModalAttention_8237747274139
// MI455X (gfx1250) — matrix operand order repaired and checked by reading, not compiled, not run on hardware
//
#include <hip/hip_runtime.h>

#define D_MODEL   512
#define NUM_HEADS 8
#define D_K       64
#define SEQ       2048
#define BATCH     4

typedef __attribute__((ext_vector_type(16))) _Float16 v16h;
typedef __attribute__((ext_vector_type(8)))  _Float16 v8h;
typedef __attribute__((ext_vector_type(8)))  float    v8f;

// Build a 16-half fragment from two 16-byte runs (ds_load_b128 / global b128).
__device__ inline v16h frag2(const _Float16* p0, const _Float16* p1) {
    v8h lo = *(const v8h*)p0;
    v8h hi = *(const v8h*)p1;
    v16h r;
#pragma unroll
    for (int i = 0; i < 8; ++i) { r[i] = lo[i]; r[i + 8] = hi[i]; }
    return r;
}

__device__ inline v8f wmma16(v16h a, v16h b, v8f c) {
    return __builtin_amdgcn_wmma_f32_16x16x32_f16(false, a, false, b,
                                                  (short)0, c, false, false);
}

// Low 32 bits of a flat pointer to __shared__ = LDS byte address.
__device__ inline unsigned ldsAddr(const void* p) { return (unsigned)(size_t)p; }

// gfx1250 async DMA: 16B global -> LDS, tracked by ASYNCcnt.
__device__ inline void async_copy_b128(unsigned lds, const _Float16* g) {
    asm volatile("global_load_async_to_lds_b128 %0, %1, off"
                 :: "v"(lds), "v"((unsigned long long)(size_t)g)
                 : "memory");
}
__device__ inline void wait_async0() {
    asm volatile("s_wait_asynccnt 0" ::: "memory");
}

// ---------------------------------------------------------------------------
// Kernel 1: P = f16( X(8192x512) @ W(512x512) + bias ).
// vmode==0: head-major [B][H][S][64]   (Q, K)
// vmode==1: head-transposed [B][H][64][S]  (V, so attention can b128-copy Vt)
// Block tile 128x64, 8 waves (4x2), wave tile 32x32.  Global->reg prefetch is
// software-pipelined against the WMMAs.
// ---------------------------------------------------------------------------
__global__ __launch_bounds__(256) void proj_kernel(
    const float* __restrict__ X, const float* __restrict__ W,
    const float* __restrict__ bias, _Float16* __restrict__ Out, int vmode) {
    __shared__ _Float16 Al[128][40];
    __shared__ _Float16 Bl[64][40];

    const int t = threadIdx.x;
    const int lane = t & 31, wv = t >> 5;
    const int wm = wv >> 1, wn = wv & 1;
    const int m0 = blockIdx.y * 128;
    const int n0 = blockIdx.x * 64;
    const int mloc = lane & 15, hi = lane >> 4;

    // per-thread staging coordinates
    const int arr = t >> 3, ac4 = (t & 7) * 4;   // A: rows arr+32u, cols ac4..+3
    const int bkk = t >> 4, bn4 = (t & 15) * 4;  // B: k rows bkk+16u, n cols bn4..+3

    float4 fa[4], fb[2];
    v8f acc[2][2] = {};

    auto loadAB = [&](int k0) {
#pragma unroll
        for (int u = 0; u < 4; ++u)
            fa[u] = *(const float4*)(X + (size_t)(m0 + arr + u * 32) * D_MODEL + k0 + ac4);
#pragma unroll
        for (int u = 0; u < 2; ++u)
            fb[u] = *(const float4*)(W + (size_t)(k0 + bkk + u * 16) * D_MODEL + n0 + bn4);
    };

    loadAB(0);
    for (int k0 = 0; k0 < D_MODEL; k0 += 32) {
        __syncthreads();
#pragma unroll
        for (int u = 0; u < 4; ++u) {
            _Float16* d = &Al[arr + u * 32][ac4];
            d[0] = (_Float16)fa[u].x; d[1] = (_Float16)fa[u].y;
            d[2] = (_Float16)fa[u].z; d[3] = (_Float16)fa[u].w;
        }
#pragma unroll
        for (int u = 0; u < 2; ++u) {
            int kk = bkk + u * 16;
            Bl[bn4 + 0][kk] = (_Float16)fb[u].x; Bl[bn4 + 1][kk] = (_Float16)fb[u].y;
            Bl[bn4 + 2][kk] = (_Float16)fb[u].z; Bl[bn4 + 3][kk] = (_Float16)fb[u].w;
        }
        __syncthreads();
        if (k0 + 32 < D_MODEL) loadAB(k0 + 32);  // overlap with WMMAs below

#pragma unroll
        for (int mt = 0; mt < 2; ++mt) {
            const _Float16* ap = &Al[wm * 32 + mt * 16 + mloc][hi * 8];
            v16h a = frag2(ap, ap + 16);
#pragma unroll
            for (int nt = 0; nt < 2; ++nt) {
                const _Float16* bp = &Bl[wn * 32 + nt * 16 + mloc][hi * 8];
                v16h b = frag2(bp, bp + 16);
                acc[mt][nt] = wmma16(a, b, acc[mt][nt]);
            }
        }
    }

    const size_t dStr = vmode ? (size_t)SEQ : 1;
    const size_t sStr = vmode ? 1 : (size_t)D_K;
#pragma unroll
    for (int mt = 0; mt < 2; ++mt) {
#pragma unroll
        for (int nt = 0; nt < 2; ++nt) {
            int col = n0 + wn * 32 + nt * 16 + mloc;
            int h = col >> 6, d = col & 63;
            float bb = bias[col];
            size_t headBase = ((size_t)0 * NUM_HEADS + h) * (size_t)(D_K * SEQ);
#pragma unroll
            for (int r = 0; r < 8; ++r) {
                int row = m0 + wm * 32 + mt * 16 + r + 8 * hi;
                int b = row >> 11, s = row & (SEQ - 1);
                Out[((size_t)b * NUM_HEADS) * (size_t)(D_K * SEQ) + headBase +
                    (size_t)d * dStr + (size_t)s * sStr] =
                    (_Float16)(acc[mt][nt][r] + bb);
            }
        }
    }
}

// ---------------------------------------------------------------------------
// Kernel 2: flash attention. grid = (Sq/128, B*H). 8 waves; wave = 16 q-rows.
// K [sk][dk] and V^T [dk][sk] chunks double-buffered in LDS via async DMA
// (GLOBAL_LOAD_ASYNC_TO_LDS_B128 + s_wait_asynccnt), overlapped with WMMAs.
// ---------------------------------------------------------------------------
__global__ __launch_bounds__(256) void attn_kernel(
    const _Float16* __restrict__ Qp, const _Float16* __restrict__ Kp,
    const _Float16* __restrict__ VpT, _Float16* __restrict__ Ctx) {
    __shared__ _Float16 Kl[2][64][72];     // [buf][sk][dk]
    __shared__ _Float16 Vt[2][64][72];     // [buf][dk][sk]
    __shared__ _Float16 Pl[8][16][72];     // per-wave P tile [m][sk]

    const int t = threadIdx.x;
    const int lane = t & 31, w = t >> 5;
    const int mloc = lane & 15, hi = lane >> 4;
    const int bh = blockIdx.y;
    const size_t bhOff = (size_t)bh * SEQ * D_K;
    const _Float16* Kbase = Kp + bhOff;
    const _Float16* Vbase = VpT + bhOff;   // [dk][S] per head
    const int q0 = blockIdx.x * 128 + w * 16;

    // staging coordinates: 512 16B-chunks per matrix, 2 per thread
    const int r0 = t >> 3, c8 = (t & 7) * 8;

    auto stage = [&](int kb, int buf) {
        const _Float16* gK = Kbase + (size_t)kb * D_K;
        const _Float16* gV = Vbase + kb;
        async_copy_b128(ldsAddr(&Kl[buf][r0][c8]),      gK + (size_t)r0 * D_K + c8);
        async_copy_b128(ldsAddr(&Kl[buf][r0 + 32][c8]), gK + (size_t)(r0 + 32) * D_K + c8);
        async_copy_b128(ldsAddr(&Vt[buf][r0][c8]),      gV + (size_t)r0 * SEQ + c8);
        async_copy_b128(ldsAddr(&Vt[buf][r0 + 32][c8]), gV + (size_t)(r0 + 32) * SEQ + c8);
    };

    // Q fragments (16x64 = two 16x32 A-frags), straight from global
    const _Float16* qb = Qp + bhOff + (size_t)(q0 + mloc) * D_K;
    v16h qa0 = frag2(qb + hi * 8, qb + hi * 8 + 16);
    v16h qa1 = frag2(qb + 32 + hi * 8, qb + 32 + hi * 8 + 16);

    float mrow[8], lrow[8];
#pragma unroll
    for (int r = 0; r < 8; ++r) { mrow[r] = -1e30f; lrow[r] = 0.f; }
    v8f o[4] = {};

    stage(0, 0);
    for (int kb = 0; kb < SEQ; kb += 64) {
        const int cur = (kb >> 6) & 1;
        wait_async0();          // this wave's async copies for buffer `cur` done
        __syncthreads();        // everyone's copies done; prev buffer fully read
        if (kb + 64 < SEQ) stage(kb + 64, cur ^ 1);  // DMA next chunk now

        const _Float16 (*Kc)[72] = Kl[cur];
        const _Float16 (*Vc)[72] = Vt[cur];

        // S = Q @ K^T  (16 x 64 scores per wave)
        v8f s[4] = {};
#pragma unroll
        for (int nt = 0; nt < 4; ++nt) {
            const _Float16* bp0 = &Kc[nt * 16 + mloc][hi * 8];
            const _Float16* bp1 = &Kc[nt * 16 + mloc][32 + hi * 8];
            v16h b0 = frag2(bp0, bp0 + 16);
            v16h b1 = frag2(bp1, bp1 + 16);
            s[nt] = wmma16(qa0, b0, s[nt]);
            s[nt] = wmma16(qa1, b1, s[nt]);
        }
#pragma unroll
        for (int nt = 0; nt < 4; ++nt)
#pragma unroll
            for (int r = 0; r < 8; ++r) s[nt][r] *= 0.125f;  // 1/sqrt(64)

        // Online softmax; rows live in 16-lane halves
#pragma unroll
        for (int r = 0; r < 8; ++r) {
            float mx = -1e30f;
#pragma unroll
            for (int nt = 0; nt < 4; ++nt) mx = fmaxf(mx, s[nt][r]);
#pragma unroll
            for (int off = 1; off < 16; off <<= 1)
                mx = fmaxf(mx, __shfl_xor(mx, off, 32));
            float mn = fmaxf(mrow[r], mx);
            float alpha = __expf(mrow[r] - mn);
            mrow[r] = mn;
            float rs = 0.f;
#pragma unroll
            for (int nt = 0; nt < 4; ++nt) {
                float p = __expf(s[nt][r] - mn);
                s[nt][r] = p;
                rs += p;
            }
#pragma unroll
            for (int off = 1; off < 16; off <<= 1)
                rs += __shfl_xor(rs, off, 32);
            lrow[r] = lrow[r] * alpha + rs;
#pragma unroll
            for (int nt = 0; nt < 4; ++nt) o[nt][r] *= alpha;
        }

        // P -> wave-private LDS (C-layout -> A-frag layout). Same-wave LDS ops
        // are in-order; fence stops compiler reordering, no barrier needed.
#pragma unroll
        for (int nt = 0; nt < 4; ++nt)
#pragma unroll
            for (int r = 0; r < 8; ++r)
                Pl[w][r + 8 * hi][nt * 16 + mloc] = (_Float16)s[nt][r];
        asm volatile("" ::: "memory");

        const _Float16* pb = &Pl[w][mloc][0];
        v16h pa0 = frag2(pb + hi * 8, pb + hi * 8 + 16);
        v16h pa1 = frag2(pb + 32 + hi * 8, pb + 32 + hi * 8 + 16);

        // O += P @ V
#pragma unroll
        for (int nt = 0; nt < 4; ++nt) {
            const _Float16* bp0 = &Vc[nt * 16 + mloc][hi * 8];
            const _Float16* bp1 = &Vc[nt * 16 + mloc][32 + hi * 8];
            v16h b0 = frag2(bp0, bp0 + 16);
            v16h b1 = frag2(bp1, bp1 + 16);
            o[nt] = wmma16(pa0, b0, o[nt]);
            o[nt] = wmma16(pa1, b1, o[nt]);
        }
    }

    // Normalize and store context as [B][S][h*64+dk] f16
    const int b = bh >> 3, h = bh & 7;
#pragma unroll
    for (int nt = 0; nt < 4; ++nt)
#pragma unroll
        for (int r = 0; r < 8; ++r) {
            int srow = q0 + r + 8 * hi;
            int col = h * D_K + nt * 16 + mloc;
            Ctx[((size_t)b * SEQ + srow) * D_MODEL + col] =
                (_Float16)(o[nt][r] / lrow[r]);
        }
}

// ---------------------------------------------------------------------------
// Kernel 3: out = LN( Ctx @ Wo + bo + residual ).  16 rows per block so the
// whole 512-wide row is resident for the LayerNorm epilogue.  Wo chunk is
// prefetched into registers while WMMAs run.
// ---------------------------------------------------------------------------
__global__ __launch_bounds__(256) void outproj_ln_kernel(
    const _Float16* __restrict__ Ctx, const float* __restrict__ Wo,
    const float* __restrict__ bo, const float* __restrict__ resid,
    const float* __restrict__ g, const float* __restrict__ beta,
    float* __restrict__ Out) {
    __shared__ _Float16 Al[16][40];
    __shared__ _Float16 Bt[512][40];      // Wo chunk, [n][k]
    __shared__ float    Xl[16][D_MODEL];  // pre-LN rows
    __shared__ float    red1[16][17], red2[16][17], mu[16], rsd[16];

    const int t = threadIdx.x;
    const int lane = t & 31, w = t >> 5;
    const int mloc = lane & 15, hi = lane >> 4;
    const int m0 = blockIdx.x * 16;

    const int ar = t >> 2, ac8 = (t & 3) * 8;        // A staging (t<64)
    const int bk0 = t >> 7, bn4 = (t & 127) * 4;     // B staging: kk = bk0+2u

    v8f acc[4] = {};
    v8h pa{};
    float4 pb[16];

    auto loadAB = [&](int k0) {
        if (t < 64)
            pa = *(const v8h*)(Ctx + (size_t)(m0 + ar) * D_MODEL + k0 + ac8);
#pragma unroll
        for (int u = 0; u < 16; ++u)
            pb[u] = *(const float4*)(Wo + (size_t)(k0 + bk0 + 2 * u) * D_MODEL + bn4);
    };

    loadAB(0);
    for (int k0 = 0; k0 < D_MODEL; k0 += 32) {
        __syncthreads();
        if (t < 64) *(v8h*)&Al[ar][ac8] = pa;
#pragma unroll
        for (int u = 0; u < 16; ++u) {
            int kk = bk0 + 2 * u;
            Bt[bn4 + 0][kk] = (_Float16)pb[u].x; Bt[bn4 + 1][kk] = (_Float16)pb[u].y;
            Bt[bn4 + 2][kk] = (_Float16)pb[u].z; Bt[bn4 + 3][kk] = (_Float16)pb[u].w;
        }
        __syncthreads();
        if (k0 + 32 < D_MODEL) loadAB(k0 + 32);  // overlap with WMMAs below

        const _Float16* ap = &Al[mloc][hi * 8];
        v16h a = frag2(ap, ap + 16);
#pragma unroll
        for (int nt = 0; nt < 4; ++nt) {
            const _Float16* bp = &Bt[w * 64 + nt * 16 + mloc][hi * 8];
            v16h b = frag2(bp, bp + 16);
            acc[nt] = wmma16(a, b, acc[nt]);
        }
    }

    // bias + residual into Xl
#pragma unroll
    for (int nt = 0; nt < 4; ++nt)
#pragma unroll
        for (int r = 0; r < 8; ++r) {
            int row = m0 + r + 8 * hi;
            int col = w * 64 + nt * 16 + mloc;
            Xl[r + 8 * hi][col] =
                acc[nt][r] + bo[col] + resid[(size_t)row * D_MODEL + col];
        }
    __syncthreads();

    // LayerNorm: 16 threads per row
    const int r = t >> 4, j = t & 15;
    float s1 = 0.f, s2 = 0.f;
    for (int c = j * 32; c < j * 32 + 32; ++c) {
        float v = Xl[r][c];
        s1 += v; s2 += v * v;
    }
    red1[r][j] = s1; red2[r][j] = s2;
    __syncthreads();
    if (j == 0) {
        float a = 0.f, b2 = 0.f;
        for (int jj = 0; jj < 16; ++jj) { a += red1[r][jj]; b2 += red2[r][jj]; }
        float mean = a * (1.0f / D_MODEL);
        float var = b2 * (1.0f / D_MODEL) - mean * mean;
        mu[r] = mean;
        rsd[r] = rsqrtf(var + 1e-5f);
    }
    __syncthreads();
    const int orow = m0 + r;
    for (int c = j * 32; c < j * 32 + 32; ++c)
        Out[(size_t)orow * D_MODEL + c] = (Xl[r][c] - mu[r]) * rsd[r] * g[c] + beta[c];
}

// ---------------------------------------------------------------------------
extern "C" void kernel_launch(void* const* d_in, const int* in_sizes, int n_in,
                              void* d_out, int out_size, void* d_ws, size_t ws_size,
                              hipStream_t stream) {
    const float* query = (const float*)d_in[0];
    const float* key   = (const float*)d_in[1];
    const float* value = (const float*)d_in[2];
    const float* wq = (const float*)d_in[3];
    const float* bq = (const float*)d_in[4];
    const float* wk = (const float*)d_in[5];
    const float* bk = (const float*)d_in[6];
    const float* wv = (const float*)d_in[7];
    const float* bv = (const float*)d_in[8];
    const float* wo = (const float*)d_in[9];
    const float* bo = (const float*)d_in[10];
    const float* lng = (const float*)d_in[11];
    const float* lnb = (const float*)d_in[12];

    _Float16* ws = (_Float16*)d_ws;
    const size_t seg = (size_t)BATCH * NUM_HEADS * SEQ * D_K;  // 4,194,304 halves
    _Float16* Qp  = ws;
    _Float16* Kp  = ws + seg;
    _Float16* Vp  = ws + 2 * seg;   // stored [B][H][64][S]
    _Float16* Ctx = ws + 3 * seg;

    dim3 blk(256);
    proj_kernel<<<dim3(8, 64), blk, 0, stream>>>(query, wq, bq, Qp, 0);
    proj_kernel<<<dim3(8, 64), blk, 0, stream>>>(key,   wk, bk, Kp, 0);
    proj_kernel<<<dim3(8, 64), blk, 0, stream>>>(value, wv, bv, Vp, 1);
    attn_kernel<<<dim3(16, 32), blk, 0, stream>>>(Qp, Kp, Vp, Ctx);
    outproj_ln_kernel<<<512, blk, 0, stream>>>(Ctx, wo, bo, query, lng, lnb,
                                               (float*)d_out);
}
